// DGCNN_64166811402456
// MI455X (gfx1250) — compile-verified, matrix operand order checked by reading, not run on hardware
//
#include <hip/hip_runtime.h>
#include <cstdint>

typedef __attribute__((ext_vector_type(16))) _Float16 v16h;
typedef __attribute__((ext_vector_type(8)))  _Float16 v8h;
typedef __attribute__((ext_vector_type(8)))  float    v8f;

#define BATCH 8
#define NPTS  2048
#define KNN   20
#define NEGS  0.2f
#define EPSW  1e-5f

// ---------------- utility ----------------
__global__ void zero_f32(float* __restrict__ p, int n) {
  int i = blockIdx.x * blockDim.x + threadIdx.x;
  if (i < n) p[i] = 0.f;
}

// x (B,3,N) -> pts (B,N,3)
__global__ void transpose_in(const float* __restrict__ x, float* __restrict__ pts) {
  int gid = blockIdx.x * blockDim.x + threadIdx.x;
  const int total = BATCH * NPTS * 3;
  if (gid >= total) return;
  int c = gid % 3;
  int n = (gid / 3) % NPTS;
  int b = gid / (3 * NPTS);
  pts[gid] = x[(size_t)b * 3 * NPTS + (size_t)c * NPTS + n];
}

// pack W f32 [Cout,Cin] -> f16 [CoutP, CinP] zero-padded
__global__ void pack_w_f16(const float* __restrict__ W, _Float16* __restrict__ Wp,
                           int Cout, int Cin, int CoutP, int CinP) {
  int gid = blockIdx.x * blockDim.x + threadIdx.x;
  int total = CoutP * CinP;
  if (gid >= total) return;
  int k = gid % CinP;
  int o = gid / CinP;
  float v = (o < Cout && k < Cin) ? W[(size_t)o * Cin + k] : 0.f;
  Wp[gid] = (_Float16)v;
}

// ---------------- kNN (top-K smallest squared distance, self included) ----------------
template <int C>
__global__ void knn_kernel(const float* __restrict__ X, int* __restrict__ idx) {
  __shared__ float tile[128 * C];
  const int b = blockIdx.y;
  const int q = blockIdx.x * blockDim.x + threadIdx.x; // exact grid
  const float* Xb = X + (size_t)b * NPTS * C;

  float qv[C];
#pragma unroll
  for (int c = 0; c < C; ++c) qv[c] = Xb[(size_t)q * C + c];

  float bd[KNN];
  int   bi[KNN];
#pragma unroll
  for (int j = 0; j < KNN; ++j) { bd[j] = 3.0e38f; bi[j] = 0; }

  for (int t0 = 0; t0 < NPTS; t0 += 128) {
    __syncthreads();
    for (int i = threadIdx.x; i < 128 * C; i += blockDim.x)
      tile[i] = Xb[(size_t)t0 * C + i];
    __syncthreads();
    for (int m = 0; m < 128; ++m) {
      float d = 0.f;
#pragma unroll
      for (int c = 0; c < C; ++c) {
        float t = qv[c] - tile[m * C + c];
        d += t * t;
      }
      if (d < bd[KNN - 1]) { // insertion keeps stable lower-index-first on ties
        int j = KNN - 1;
        while (j > 0 && bd[j - 1] > d) {
          bd[j] = bd[j - 1]; bi[j] = bi[j - 1]; --j;
        }
        bd[j] = d; bi[j] = t0 + m;
      }
    }
  }
  int* op = idx + ((size_t)b * NPTS + q) * KNN;
  for (int j = 0; j < KNN; ++j) op[j] = bi[j];
}

// ---------------- edge feature gather: e = [xi, x_nbr - xi], f16, zero-padded to ldE ----------------
__global__ void gather_edge(const float* __restrict__ X, const int* __restrict__ idx,
                            _Float16* __restrict__ E, int C, int ldE) {
  int gid = blockIdx.x * blockDim.x + threadIdx.x;
  const int total = BATCH * NPTS * KNN;
  if (gid >= total) return;
  int n  = (gid / KNN) % NPTS;
  int b  = gid / (KNN * NPTS);
  int nb = idx[gid];
  const float* xi = X + ((size_t)b * NPTS + n) * C;
  const float* xn = X + ((size_t)b * NPTS + nb) * C;
  _Float16* e = E + (size_t)gid * ldE;
  for (int c = 0; c < C; ++c) {
    float a = xi[c];
    e[c]     = (_Float16)a;
    e[C + c] = (_Float16)(xn[c] - a);
  }
  for (int c = 2 * C; c < ldE; ++c) e[c] = (_Float16)0.f;
}

// ---------------- WMMA GEMM: Z[rows,ldz] = A[rows,lda](f16) @ Wp[CoutP,ldw]^T ----------------
// Each wave computes 4 row tiles (64 rows) x 16 cols, reusing the B fragment 4x.
// Requirements (arranged by host): lda % 32 == 0, ldw % 32 == 0 (padded Cin),
// gridDim.y * 16 == CoutP, rows % 512 == 0 per full block. No divergence in hot loop.
__global__ void gemm_wmma(const _Float16* __restrict__ A, int lda,
                          const _Float16* __restrict__ Wp, int ldw,
                          float* __restrict__ Z, int ldz,
                          float* __restrict__ stats, int statC, int rows) {
  const int wave = threadIdx.x >> 5;
  const int lane = threadIdx.x & 31;
  const int m0 = (blockIdx.x * 8 + wave) * 64; // 4 x 16-row tiles per wave
  if (m0 >= rows) return; // wave-uniform
  const int l15 = lane & 15;
  const int h   = lane >> 4;
  const int ncol = blockIdx.y * 16 + l15;

  // Fragment layout (ISA): lane l, element j -> K = k0 + (j>>3)*16 + (l>>4)*8 + (j&7)
  // => per lane: two contiguous 8-half (16B) chunks at k0 + h*8 and k0 + 16 + h*8.
  const _Float16* Bp = Wp + (size_t)ncol * ldw + (h << 3);
  const _Float16* Ap[4];
#pragma unroll
  for (int t = 0; t < 4; ++t)
    Ap[t] = A + (size_t)(m0 + t * 16 + l15) * lda + (h << 3);

  v8f acc[4] = {{}, {}, {}, {}};
  for (int k0 = 0; k0 < ldw; k0 += 32) {
    v8h b0 = *(const v8h*)(Bp + k0);
    v8h b1 = *(const v8h*)(Bp + k0 + 16);
    v16h bf = __builtin_shufflevector(b0, b1, 0, 1, 2, 3, 4, 5, 6, 7,
                                              8, 9, 10, 11, 12, 13, 14, 15);
#pragma unroll
    for (int t = 0; t < 4; ++t) {
      __builtin_prefetch(Ap[t] + k0 + 32, 0, 0); // global_prefetch_b8, next chunk
      v8h a0 = *(const v8h*)(Ap[t] + k0);
      v8h a1 = *(const v8h*)(Ap[t] + k0 + 16);
      v16h af = __builtin_shufflevector(a0, a1, 0, 1, 2, 3, 4, 5, 6, 7,
                                                8, 9, 10, 11, 12, 13, 14, 15);
      acc[t] = __builtin_amdgcn_wmma_f32_16x16x32_f16(
          /*neg_a=*/false, af, /*neg_b=*/false, bf,
          /*c_mod=*/(short)0, acc[t], /*reuse_a=*/false, /*reuse_b=*/false);
    }
  }

  float s = 0.f, s2 = 0.f;
#pragma unroll
  for (int t = 0; t < 4; ++t) {
#pragma unroll
    for (int i = 0; i < 8; ++i) {
      int m = m0 + t * 16 + (h << 3) + i; // C/D layout: VGPR i, lanes 16-31 +8 rows
      float v = acc[t][i];
      Z[(size_t)m * ldz + ncol] = v;
      s += v; s2 += v * v;
    }
  }
  if (stats) {
    atomicAdd(&stats[ncol], s);
    atomicAdd(&stats[statC + ncol], s2);
  }
}

// ---------------- BN (batch stats) + LeakyReLU -> f16 ----------------
__global__ void bn_lrelu_to_f16(const float* __restrict__ Z, const float* __restrict__ stats,
                                const float* __restrict__ g, const float* __restrict__ bta,
                                _Float16* __restrict__ out, int C, size_t rows) {
  size_t gid = (size_t)blockIdx.x * blockDim.x + threadIdx.x;
  size_t total = rows * (size_t)C;
  if (gid >= total) return;
  int c = (int)(gid % C);
  float cnt  = (float)rows;
  float mean = stats[c] / cnt;
  float var  = stats[C + c] / cnt - mean * mean;
  float sc   = g[c] * rsqrtf(var + EPSW);
  float sh   = bta[c] - mean * sc;
  float v = Z[gid] * sc + sh;
  v = v >= 0.f ? v : NEGS * v;
  out[gid] = (_Float16)v;
}

// ---------------- BN + LeakyReLU + max over K neighbors -> f32 [B*N,64] ----------------
__global__ void bn_lrelu_maxk(const float* __restrict__ Z, const float* __restrict__ stats,
                              const float* __restrict__ g, const float* __restrict__ bta,
                              float* __restrict__ xout) {
  int gid = blockIdx.x * blockDim.x + threadIdx.x;
  const int total = BATCH * NPTS * 64;
  if (gid >= total) return;
  int c  = gid % 64;
  int pn = gid / 64;
  float cnt  = (float)(BATCH * NPTS * KNN);
  float mean = stats[c] / cnt;
  float var  = stats[64 + c] / cnt - mean * mean;
  float sc   = g[c] * rsqrtf(var + EPSW);
  float sh   = bta[c] - mean * sc;
  float mx = -3.0e38f;
  for (int k = 0; k < KNN; ++k) {
    float v = Z[((size_t)pn * KNN + k) * 64 + c] * sc + sh;
    v = v >= 0.f ? v : NEGS * v;
    mx = fmaxf(mx, v);
  }
  xout[gid] = mx;
}

// ---------------- concat [x1|x2|x3] -> f16 [B*N,192] ----------------
__global__ void build_hc(const float* __restrict__ x1, const float* __restrict__ x2,
                         const float* __restrict__ x3, _Float16* __restrict__ hc) {
  int gid = blockIdx.x * blockDim.x + threadIdx.x;
  const int total = BATCH * NPTS * 192;
  if (gid >= total) return;
  int c  = gid % 192;
  int pn = gid / 192;
  float v = (c < 64)  ? x1[(size_t)pn * 64 + c]
          : (c < 128) ? x2[(size_t)pn * 64 + (c - 64)]
                      : x3[(size_t)pn * 64 + (c - 128)];
  hc[gid] = (_Float16)v;
}

// ---------------- BN + LeakyReLU + max over N -> gmax [B,1024] ----------------
__global__ void bn_lrelu_gmax(const float* __restrict__ Z4, const float* __restrict__ stats,
                              const float* __restrict__ g, const float* __restrict__ bta,
                              float* __restrict__ gmax) {
  int gid = blockIdx.x * blockDim.x + threadIdx.x;
  if (gid >= BATCH * 1024) return;
  int c = gid % 1024;
  int b = gid / 1024;
  float cnt  = (float)(BATCH * NPTS);
  float mean = stats[c] / cnt;
  float var  = stats[1024 + c] / cnt - mean * mean;
  float sc   = g[c] * rsqrtf(var + EPSW);
  float sh   = bta[c] - mean * sc;
  float mx = -3.0e38f;
  for (int n = 0; n < NPTS; ++n) {
    float v = Z4[((size_t)b * NPTS + n) * 1024 + c] * sc + sh;
    v = v >= 0.f ? v : NEGS * v;
    mx = fmaxf(mx, v);
  }
  gmax[gid] = mx;
}

// ---------------- concat [gmax|x1|x2|x3] -> f16 [B*N,1216] ----------------
__global__ void build_h5in(const float* __restrict__ gmax, const float* __restrict__ x1,
                           const float* __restrict__ x2, const float* __restrict__ x3,
                           _Float16* __restrict__ hh) {
  size_t gid = (size_t)blockIdx.x * blockDim.x + threadIdx.x;
  const size_t total = (size_t)BATCH * NPTS * 1216;
  if (gid >= total) return;
  int c  = (int)(gid % 1216);
  int pn = (int)(gid / 1216);
  int b  = pn / NPTS;
  float v;
  if (c < 1024) {
    v = gmax[(size_t)b * 1024 + c];
  } else {
    int c2 = c - 1024;
    v = (c2 < 64)  ? x1[(size_t)pn * 64 + c2]
      : (c2 < 128) ? x2[(size_t)pn * 64 + (c2 - 64)]
                   : x3[(size_t)pn * 64 + (c2 - 128)];
  }
  hh[gid] = (_Float16)v;
}

// ---------------- final bias + transpose (B,N,64pad) -> (B,50,N) ----------------
__global__ void bias_transpose_out(const float* __restrict__ Zf, const float* __restrict__ bfv,
                                   float* __restrict__ out) {
  int gid = blockIdx.x * blockDim.x + threadIdx.x;
  const int total = BATCH * 50 * NPTS;
  if (gid >= total) return;
  int n = gid % NPTS;
  int o = (gid / NPTS) % 50;
  int b = gid / (50 * NPTS);
  out[gid] = Zf[((size_t)b * NPTS + n) * 64 + o] + bfv[o];
}

extern "C" void kernel_launch(void* const* d_in, const int* in_sizes, int n_in,
                              void* d_out, int out_size, void* d_ws, size_t ws_size,
                              hipStream_t stream) {
  (void)in_sizes; (void)n_in; (void)out_size; (void)ws_size;
  const float* x   = (const float*)d_in[0];
  const float* w1a = (const float*)d_in[1];  const float* g1a = (const float*)d_in[2];  const float* b1a = (const float*)d_in[3];
  const float* w1b = (const float*)d_in[4];  const float* g1b = (const float*)d_in[5];  const float* b1b = (const float*)d_in[6];
  const float* w2a = (const float*)d_in[7];  const float* g2a = (const float*)d_in[8];  const float* b2a = (const float*)d_in[9];
  const float* w2b = (const float*)d_in[10]; const float* g2b = (const float*)d_in[11]; const float* b2b = (const float*)d_in[12];
  const float* w3  = (const float*)d_in[13]; const float* g3  = (const float*)d_in[14]; const float* b3  = (const float*)d_in[15];
  const float* w4  = (const float*)d_in[16]; const float* g4  = (const float*)d_in[17]; const float* b4  = (const float*)d_in[18];
  const float* w5  = (const float*)d_in[19]; const float* g5  = (const float*)d_in[20]; const float* b5  = (const float*)d_in[21];
  const float* w6  = (const float*)d_in[22]; const float* g6  = (const float*)d_in[23]; const float* b6  = (const float*)d_in[24];
  const float* w7  = (const float*)d_in[25]; const float* g7  = (const float*)d_in[26]; const float* b7  = (const float*)d_in[27];
  const float* wf  = (const float*)d_in[28]; const float* bfv = (const float*)d_in[29];

  const size_t EDGE_ROWS = (size_t)BATCH * NPTS * KNN;  // 327680 (mult of 512)
  const size_t PT_ROWS   = (size_t)BATCH * NPTS;        // 16384  (mult of 512)

  char* ws = (char*)d_ws;
  size_t off = 0;
  auto carve = [&](size_t bytes) -> char* {
    char* p = ws + off;
    off += (bytes + 255) & ~(size_t)255;
    return p;
  };
  float*    pts   = (float*)carve(sizeof(float) * BATCH * NPTS * 3);
  float*    x1    = (float*)carve(sizeof(float) * BATCH * NPTS * 64);
  float*    x2    = (float*)carve(sizeof(float) * BATCH * NPTS * 64);
  float*    x3    = (float*)carve(sizeof(float) * BATCH * NPTS * 64);
  int*      idx   = (int*)carve(sizeof(int) * BATCH * NPTS * KNN);
  _Float16* bufA  = (_Float16*)carve(sizeof(_Float16) * EDGE_ROWS * 128); // edge feats / hc / h5in / h6
  float*    bufB  = (float*)carve(sizeof(float) * EDGE_ROWS * 64);        // pre-activations
  _Float16* bufC  = (_Float16*)carve(sizeof(_Float16) * EDGE_ROWS * 64);  // activated f16 / h5 / h7
  _Float16* Wp    = (_Float16*)carve(sizeof(_Float16) * 512 * 1216);      // packed weights (max w5)
  float*    stats = (float*)carve(sizeof(float) * 2048);
  float*    gmaxb = (float*)carve(sizeof(float) * BATCH * 1024);

  // A: f16 [rows, lda]; W: f32 [Cout, Cin]; Z: f32 [rows, ldz]; CoutP = padded cols.
  auto launch_gemm = [&](const _Float16* A, int lda, const float* W, int Cin, int Cout,
                         float* Z, int ldz, int CoutP, float* st, size_t rows) {
    int CinP = (Cin + 31) & ~31;
    pack_w_f16<<<(CoutP * CinP + 255) / 256, 256, 0, stream>>>(W, Wp, Cout, Cin, CoutP, CinP);
    if (st) zero_f32<<<(2 * CoutP + 255) / 256, 256, 0, stream>>>(st, 2 * CoutP);
    dim3 grid((unsigned)((rows + 511) / 512), (unsigned)(CoutP / 16));
    gemm_wmma<<<grid, 256, 0, stream>>>(A, lda, Wp, CinP, Z, ldz, st, CoutP, (int)rows);
  };

  // ---- edge conv 1 (points, C=3, edge dim 6 padded to 32) ----
  transpose_in<<<(BATCH * NPTS * 3 + 255) / 256, 256, 0, stream>>>(x, pts);
  knn_kernel<3><<<dim3(NPTS / 256, BATCH), 256, 0, stream>>>(pts, idx);
  gather_edge<<<(unsigned)((EDGE_ROWS + 255) / 256), 256, 0, stream>>>(pts, idx, bufA, 3, 32);
  launch_gemm(bufA, 32, w1a, 6, 64, bufB, 64, 64, stats, EDGE_ROWS);
  bn_lrelu_to_f16<<<(unsigned)((EDGE_ROWS * 64 + 255) / 256), 256, 0, stream>>>(bufB, stats, g1a, b1a, bufC, 64, EDGE_ROWS);
  launch_gemm(bufC, 64, w1b, 64, 64, bufB, 64, 64, stats, EDGE_ROWS);
  bn_lrelu_maxk<<<(unsigned)((PT_ROWS * 64 + 255) / 256), 256, 0, stream>>>(bufB, stats, g1b, b1b, x1);

  // ---- edge conv 2 (C=64, edge dim 128) ----
  knn_kernel<64><<<dim3(NPTS / 256, BATCH), 256, 0, stream>>>(x1, idx);
  gather_edge<<<(unsigned)((EDGE_ROWS + 255) / 256), 256, 0, stream>>>(x1, idx, bufA, 64, 128);
  launch_gemm(bufA, 128, w2a, 128, 64, bufB, 64, 64, stats, EDGE_ROWS);
  bn_lrelu_to_f16<<<(unsigned)((EDGE_ROWS * 64 + 255) / 256), 256, 0, stream>>>(bufB, stats, g2a, b2a, bufC, 64, EDGE_ROWS);
  launch_gemm(bufC, 64, w2b, 64, 64, bufB, 64, 64, stats, EDGE_ROWS);
  bn_lrelu_maxk<<<(unsigned)((PT_ROWS * 64 + 255) / 256), 256, 0, stream>>>(bufB, stats, g2b, b2b, x2);

  // ---- edge conv 3 (C=64, edge dim 128, single layer) ----
  knn_kernel<64><<<dim3(NPTS / 256, BATCH), 256, 0, stream>>>(x2, idx);
  gather_edge<<<(unsigned)((EDGE_ROWS + 255) / 256), 256, 0, stream>>>(x2, idx, bufA, 64, 128);
  launch_gemm(bufA, 128, w3, 128, 64, bufB, 64, 64, stats, EDGE_ROWS);
  bn_lrelu_maxk<<<(unsigned)((PT_ROWS * 64 + 255) / 256), 256, 0, stream>>>(bufB, stats, g3, b3, x3);

  // ---- global feature: h4 = lrelu(bn(hc @ w4^T)); gmax over N ----
  build_hc<<<(unsigned)((PT_ROWS * 192 + 255) / 256), 256, 0, stream>>>(x1, x2, x3, bufA);
  launch_gemm(bufA, 192, w4, 192, 1024, bufB, 1024, 1024, stats, PT_ROWS);
  bn_lrelu_gmax<<<(BATCH * 1024 + 255) / 256, 256, 0, stream>>>(bufB, stats, g4, b4, gmaxb);

  // ---- segmentation head ----
  build_h5in<<<(unsigned)((PT_ROWS * 1216 + 255) / 256), 256, 0, stream>>>(gmaxb, x1, x2, x3, bufA);
  launch_gemm(bufA, 1216, w5, 1216, 512, bufB, 512, 512, stats, PT_ROWS);
  bn_lrelu_to_f16<<<(unsigned)((PT_ROWS * 512 + 255) / 256), 256, 0, stream>>>(bufB, stats, g5, b5, bufC, 512, PT_ROWS);
  launch_gemm(bufC, 512, w6, 512, 256, bufB, 256, 256, stats, PT_ROWS);
  bn_lrelu_to_f16<<<(unsigned)((PT_ROWS * 256 + 255) / 256), 256, 0, stream>>>(bufB, stats, g6, b6, bufA, 256, PT_ROWS);
  launch_gemm(bufA, 256, w7, 256, 128, bufB, 128, 128, stats, PT_ROWS);
  bn_lrelu_to_f16<<<(unsigned)((PT_ROWS * 128 + 255) / 256), 256, 0, stream>>>(bufB, stats, g7, b7, bufC, 128, PT_ROWS);

  // ---- classifier (50 -> padded 64 cols) + bias + transpose ----
  launch_gemm(bufC, 128, wf, 128, 50, bufB, 64, 64, nullptr, PT_ROWS);
  bias_transpose_out<<<(BATCH * 50 * NPTS + 255) / 256, 256, 0, stream>>>(bufB, bfv, (float*)d_out);
}
